// GAT_4587025072840
// MI455X (gfx1250) — matrix operand order repaired and checked by reading, not compiled, not run on hardware
//
#include <hip/hip_runtime.h>
#include <hip/hip_bf16.h>

#define NN 100000
#define NE 1600000
#define NEG_SLOPE 0.2f
#define BN_EPS 1e-5f

typedef __attribute__((ext_vector_type(16))) __bf16 bf16x16;
typedef __attribute__((ext_vector_type(8)))  __bf16 bf16x8;
typedef __attribute__((ext_vector_type(8)))  float  v8f;

union BfPack { bf16x16 v; bf16x8 h[2]; };

// ---------------- conversion helpers ----------------

__global__ void k_f32_to_bf16(const float* __restrict__ x, __bf16* __restrict__ y, int n) {
    int i = blockIdx.x * blockDim.x + threadIdx.x;
    if (i < n) y[i] = (__bf16)x[i];
}

// Pack W [K][128] row-major fp32 into WMMA-B per-lane layout:
// Bp[((kt*8+nt)*32+lane)*16 + 2j + {0,1}] holds K = kt*32 + (lane>=16?16:0) + 2j (+1),
// column nt*16 + (lane&15).
__global__ void k_pack_w(const float* __restrict__ W, __bf16* __restrict__ Bp, int K) {
    int tid = blockIdx.x * blockDim.x + threadIdx.x;
    int ktiles = K >> 5;
    int total = ktiles * 8 * 32 * 8;
    if (tid >= total) return;
    int j    = tid & 7;
    int lane = (tid >> 3) & 31;
    int nt   = (tid >> 8) & 7;
    int kt   = tid >> 11;
    int col  = nt * 16 + (lane & 15);
    int kb   = kt * 32 + ((lane >> 4) << 4) + 2 * j;
    __bf16* out = Bp + ((size_t)((kt * 8 + nt) * 32 + lane)) * 16 + 2 * j;
    out[0] = (__bf16)W[(size_t)kb * 128 + col];
    out[1] = (__bf16)W[(size_t)(kb + 1) * 128 + col];
}

// ---------------- WMMA GEMM: H[M,128] = A[M,K](bf16) * Bpacked ----------------

__global__ void k_gemm_wmma(const __bf16* __restrict__ A, const __bf16* __restrict__ Bp,
                            float* __restrict__ H, int M, int K) {
    int lane = threadIdx.x & 31;
    int wave = threadIdx.x >> 5;
    int mt = blockIdx.x * (blockDim.x >> 5) + wave;
    if (mt * 16 >= M) return;                       // wave-uniform: EXEC stays all-1s
    int row = mt * 16 + (lane & 15);
    int kh  = (lane >> 4) * 16;                     // 0 or 16
    int ktiles = K >> 5;

    v8f acc[8];
#pragma unroll
    for (int i = 0; i < 8; ++i) acc[i] = (v8f){0.f,0.f,0.f,0.f,0.f,0.f,0.f,0.f};

    for (int kt = 0; kt < ktiles; ++kt) {
        const __bf16* arow = A + (size_t)row * K + kt * 32 + kh;
        BfPack a;
        a.h[0] = *(const bf16x8*)(arow);            // K = kh .. kh+7
        a.h[1] = *(const bf16x8*)(arow + 8);        // K = kh+8 .. kh+15
        const __bf16* bbase = Bp + (size_t)(kt * 8) * 32 * 16 + (size_t)lane * 16;
#pragma unroll
        for (int nt = 0; nt < 8; ++nt) {
            BfPack b;
            const __bf16* bp = bbase + (size_t)nt * 32 * 16;
            b.h[0] = *(const bf16x8*)(bp);
            b.h[1] = *(const bf16x8*)(bp + 8);
            acc[nt] = __builtin_amdgcn_wmma_f32_16x16x32_bf16(
                false, a.v, false, b.v, (short)0, acc[nt], false, false);
        }
    }

    int rbase = mt * 16 + (lane >> 4) * 8;
    int cbase = lane & 15;
#pragma unroll
    for (int nt = 0; nt < 8; ++nt)
#pragma unroll
        for (int j = 0; j < 8; ++j)
            H[(size_t)(rbase + j) * 128 + nt * 16 + cbase] = acc[nt][j];
}

// ---------------- attention logits per node ----------------

__global__ void k_al(const float* __restrict__ h, const float* __restrict__ asrc,
                     const float* __restrict__ adst,
                     float* __restrict__ als, float* __restrict__ ald) {
    int tid = blockIdx.x * blockDim.x + threadIdx.x;
    if (tid >= NN * 4) return;
    int n = tid >> 2, hd = tid & 3;
    const float* hp = h + (size_t)n * 128 + hd * 32;
    const float* as = asrc + hd * 32;
    const float* ad = adst + hd * 32;
    float s = 0.f, d = 0.f;
#pragma unroll
    for (int c = 0; c < 32; ++c) { float v = hp[c]; s += v * as[c]; d += v * ad[c]; }
    als[tid] = s; ald[tid] = d;
}

// ---------------- edge phase kernels ----------------

__global__ void k_fill_neginf(float* __restrict__ p, int n) {
    int i = blockIdx.x * blockDim.x + threadIdx.x;
    if (i < n) p[i] = -__builtin_huge_valf();
}

__device__ __forceinline__ void atomicMaxF(float* addr, float val) {
    if (val >= 0.f) atomicMax((int*)addr, __float_as_int(val));
    else            atomicMin((unsigned int*)addr, (unsigned int)__float_as_int(val));
}

__device__ __forceinline__ float edge_logit(const float* als, const float* ald,
                                            int s, int d, int hd) {
    float v = als[s * 4 + hd] + ald[d * 4 + hd];
    return v >= 0.f ? v : NEG_SLOPE * v;
}

__global__ void k_edge_max(const int* __restrict__ src, const int* __restrict__ dst,
                           const float* __restrict__ als, const float* __restrict__ ald,
                           float* __restrict__ emax) {
    int tid = blockIdx.x * blockDim.x + threadIdx.x;
    if (tid >= NE * 4) return;
    int e = tid >> 2, hd = tid & 3;
    int s = src[e], d = dst[e];
    atomicMaxF(&emax[d * 4 + hd], edge_logit(als, ald, s, d, hd));
}

__global__ void k_edge_exp(const int* __restrict__ src, const int* __restrict__ dst,
                           const float* __restrict__ als, const float* __restrict__ ald,
                           const float* __restrict__ emax,
                           float* __restrict__ denom, float* __restrict__ ee) {
    int tid = blockIdx.x * blockDim.x + threadIdx.x;
    if (tid >= NE * 4) return;
    int e = tid >> 2, hd = tid & 3;
    int s = src[e], d = dst[e];
    float x = expf(edge_logit(als, ald, s, d, hd) - emax[d * 4 + hd]);
    ee[tid] = x;
    atomicAdd(&denom[d * 4 + hd], x);
}

// wave = one edge, lane = channel within head; loops 4 heads
__global__ void k_edge_agg(const int* __restrict__ src, const int* __restrict__ dst,
                           const float* __restrict__ ee, const float* __restrict__ denom,
                           const float* __restrict__ h, float* __restrict__ accum) {
    int gt = blockIdx.x * blockDim.x + threadIdx.x;
    int e = gt >> 5;
    int c = gt & 31;
    if (e >= NE) return;
    int s = src[e], d = dst[e];
#pragma unroll
    for (int hd = 0; hd < 4; ++hd) {
        float alpha = ee[e * 4 + hd] / denom[d * 4 + hd];
        float v = h[(size_t)s * 128 + hd * 32 + c] * alpha;
        atomicAdd(&accum[(size_t)d * 128 + hd * 32 + c], v);
    }
}

// ---------------- head mean + bias, batchnorm ----------------

__global__ void k_head_mean(const float* __restrict__ accum, const float* __restrict__ bias,
                            float* __restrict__ out) {
    int tid = blockIdx.x * blockDim.x + threadIdx.x;
    if (tid >= NN * 32) return;
    int n = tid >> 5, c = tid & 31;
    const float* a = accum + (size_t)n * 128 + c;
    out[tid] = 0.25f * (a[0] + a[32] + a[64] + a[96]) + bias[c];
}

__global__ void k_bn_stats(const float* __restrict__ x, float* __restrict__ stats) {
    int c = blockIdx.x;           // 32 blocks, one channel each
    int t = threadIdx.x;          // 256 threads
    float s = 0.f, q = 0.f;
    for (int n = t; n < NN; n += 256) {
        float v = x[(size_t)n * 32 + c];
        s += v; q += v * v;
    }
    __shared__ float sh[256], shq[256];
    sh[t] = s; shq[t] = q; __syncthreads();
    for (int o = 128; o > 0; o >>= 1) {
        if (t < o) { sh[t] += sh[t + o]; shq[t] += shq[t + o]; }
        __syncthreads();
    }
    if (t == 0) { stats[c] = sh[0]; stats[32 + c] = shq[0]; }
}

__global__ void k_bn_relu_bf16(const float* __restrict__ x, const float* __restrict__ stats,
                               const float* __restrict__ g, const float* __restrict__ b,
                               __bf16* __restrict__ y) {
    int tid = blockIdx.x * blockDim.x + threadIdx.x;
    if (tid >= NN * 32) return;
    int c = tid & 31;
    float inv_n = 1.f / (float)NN;
    float mu  = stats[c] * inv_n;
    float var = stats[32 + c] * inv_n - mu * mu;
    float v = (x[tid] - mu) * rsqrtf(var + BN_EPS) * g[c] + b[c];
    y[tid] = (__bf16)(v > 0.f ? v : 0.f);
}

// ---------------- host launcher ----------------

extern "C" void kernel_launch(void* const* d_in, const int* in_sizes, int n_in,
                              void* d_out, int out_size, void* d_ws, size_t ws_size,
                              hipStream_t stream) {
    const float* x         = (const float*)d_in[0];
    const int*   src       = (const int*)d_in[1];
    const int*   dst       = ((const int*)d_in[1]) + NE;
    const float* edge_attr = (const float*)d_in[2];
    const float* W[3]    = {(const float*)d_in[3],  (const float*)d_in[9],  (const float*)d_in[15]};
    const float* asrc[3] = {(const float*)d_in[4],  (const float*)d_in[10], (const float*)d_in[16]};
    const float* adst[3] = {(const float*)d_in[5],  (const float*)d_in[11], (const float*)d_in[17]};
    const float* bias[3] = {(const float*)d_in[6],  (const float*)d_in[12], (const float*)d_in[18]};
    const float* gam[2]  = {(const float*)d_in[7],  (const float*)d_in[13]};
    const float* bet[2]  = {(const float*)d_in[8],  (const float*)d_in[14]};

    // workspace carve-up (256B aligned)
    char* ws = (char*)d_ws;
    size_t off = 0;
    auto carve = [&](size_t bytes) -> char* {
        char* p = ws + off;
        off = (off + bytes + 255) & ~(size_t)255;
        return p;
    };
    __bf16* xbf   = (__bf16*)carve((size_t)NN * 128 * 2);   // GEMM A (reused across layers)
    float*  hbuf  = (float*)carve((size_t)NN * 128 * 4);    // projected features
    float*  accum = (float*)carve((size_t)NN * 128 * 4);    // per-head aggregation
    float*  als   = (float*)carve((size_t)NN * 4 * 4);
    float*  ald   = (float*)carve((size_t)NN * 4 * 4);
    float*  emax  = (float*)carve((size_t)NN * 4 * 4);
    float*  denom = (float*)carve((size_t)NN * 4 * 4);
    float*  ee    = (float*)carve((size_t)NE * 4 * 4);
    float*  hn    = (float*)carve((size_t)NN * 32 * 4);     // head-mean result
    float*  stats = (float*)carve(64 * 4);
    __bf16* bpack = (__bf16*)carve(4 * 8 * 32 * 16 * 2);    // packed weights (max K=128)

    const int T = 256;
    float* out_h = (float*)d_out;

    // layer-0 input: x fp32 -> bf16
    k_f32_to_bf16<<<(NN * 128 + T - 1) / T, T, 0, stream>>>(x, xbf, NN * 128);

    for (int l = 0; l < 3; ++l) {
        int K = (l == 0) ? 128 : 32;
        int ktiles = K >> 5;

        // pack weights into WMMA-B layout
        int pw = ktiles * 8 * 32 * 8;
        k_pack_w<<<(pw + T - 1) / T, T, 0, stream>>>(W[l], bpack, K);

        // projection GEMM (WMMA): hbuf = xbf @ W
        int mtiles = NN / 16;                 // 6250 exactly
        k_gemm_wmma<<<(mtiles + 3) / 4, 128, 0, stream>>>(xbf, bpack, hbuf, NN, K);

        // attention logits
        k_al<<<(NN * 4 + T - 1) / T, T, 0, stream>>>(hbuf, asrc[l], adst[l], als, ald);

        // init segment buffers
        k_fill_neginf<<<(NN * 4 + T - 1) / T, T, 0, stream>>>(emax, NN * 4);
        hipMemsetAsync(denom, 0, (size_t)NN * 4 * 4, stream);
        hipMemsetAsync(accum, 0, (size_t)NN * 128 * 4, stream);

        // segment softmax + aggregation
        k_edge_max<<<(NE * 4 + T - 1) / T, T, 0, stream>>>(src, dst, als, ald, emax);
        k_edge_exp<<<(NE * 4 + T - 1) / T, T, 0, stream>>>(src, dst, als, ald, emax, denom, ee);
        k_edge_agg<<<((size_t)NE * 32 + T - 1) / T, T, 0, stream>>>(src, dst, ee, denom, hbuf, accum);

        // head mean + bias
        float* mean_out = (l == 2) ? out_h : hn;
        k_head_mean<<<(NN * 32 + T - 1) / T, T, 0, stream>>>(accum, bias[l], mean_out);

        if (l < 2) {
            // batchnorm + relu, downconvert to bf16 as next layer's GEMM input
            k_bn_stats<<<32, 256, 0, stream>>>(hn, stats);
            k_bn_relu_bf16<<<(NN * 32 + T - 1) / T, T, 0, stream>>>(hn, stats, gam[l], bet[l], xbf);
        }
    }

    // second tuple element: edge_attr passthrough
    hipMemcpyAsync(out_h + (size_t)NN * 32, edge_attr, (size_t)NE * 8 * 4,
                   hipMemcpyDeviceToDevice, stream);
}
